// T5Attention_32908039422081
// MI455X (gfx1250) — matrix operand order repaired and checked by reading, not compiled, not run on hardware
//
#include <hip/hip_runtime.h>
#include <hip/hip_bf16.h>

// ---------------------------------------------------------------------------
// T5 attention block for gfx1250 (MI455X), bf16 WMMA + fp32 accumulate.
// B=8, S=1024, D_MODEL=1024, H=16, D_KV=64, INNER=1024, NUM_BUCKETS=32.
// GEMMs register-blocked 16x64 per wave; flash attention stages K via
// async global->LDS copies (ASYNCcnt) when the toolchain exposes them.
// ---------------------------------------------------------------------------

typedef __attribute__((ext_vector_type(16))) __bf16 v16bf;
typedef __attribute__((ext_vector_type(8)))  __bf16 v8bf;
typedef __attribute__((ext_vector_type(4)))  __bf16 v4bf;
typedef __attribute__((ext_vector_type(8)))  float  v8f;
typedef __attribute__((ext_vector_type(4)))  int    v4i;

#define T5_B 8
#define T5_S 1024
#define T5_DM 1024
#define T5_H 16
#define T5_DKV 64
#define T5_INNER 1024

#if defined(__AMDGCN__) && __has_builtin(__builtin_amdgcn_global_load_async_to_lds_b128)
#define USE_ASYNC_LDS 1
#endif

static __device__ __forceinline__ void wait_async_zero() {
#if defined(USE_ASYNC_LDS)
#if __has_builtin(__builtin_amdgcn_s_wait_asynccnt)
    __builtin_amdgcn_s_wait_asynccnt(0);
#else
    asm volatile("s_wait_asynccnt 0" ::: "memory");
#endif
#endif
}

static __device__ __forceinline__ v8f wmma_bf16(v16bf a, v16bf b, v8f c) {
    return __builtin_amdgcn_wmma_f32_16x16x32_bf16(false, a, false, b, (short)0, c,
                                                   false, false);
}

// ------------------------- fp32 -> bf16 convert (x4) -----------------------
__global__ __launch_bounds__(256) void cvt_f32_bf16_kernel(
    const float* __restrict__ src, __bf16* __restrict__ dst, int n4) {
    int i = blockIdx.x * blockDim.x + threadIdx.x;
    if (i < n4) {
        const float4 v = ((const float4*)src)[i];
        v4bf o;
        o[0] = (__bf16)v.x; o[1] = (__bf16)v.y;
        o[2] = (__bf16)v.z; o[3] = (__bf16)v.w;
        ((v4bf*)dst)[i] = o;
    }
}

// ------------------------- relative position bias table --------------------
// tab[h * 2048 + (delta + 1023)] = emb[bucket(delta), h]
__global__ __launch_bounds__(256) void bias_table_kernel(
    const float* __restrict__ emb, float* __restrict__ tab) {
    int idx = blockIdx.x * blockDim.x + threadIdx.x;
    if (idx >= T5_H * 2048) return;
    int h = idx >> 11;
    int t = idx & 2047;
    int rp = t - 1023;                 // delta = k - q
    int bucket = 0;
    if (rp > 0) bucket += 16;          // bidirectional: num_buckets//2
    int arp = rp < 0 ? -rp : rp;
    int v;
    if (arp < 8) {                     // max_exact = 8
        v = arp;
    } else {
        // 8 + int( log(arp/8)/log(16) * 8 )
        float f = __logf((float)arp * 0.125f) * (8.0f / __logf(16.0f));
        v = 8 + (int)f;
        if (v > 15) v = 15;
    }
    bucket += v;
    tab[idx] = emb[bucket * T5_H + h];
}

// ------------------------- QKV projection GEMM ------------------------------
// qkv[m, n] = sum_d hidden[m, d] * w_qkv[n, d]; M=8192, N=3072, K=1024.
// One wave -> 16x64 tile (4 accumulators, shared A operand).
// Epilogue scatters into Q/K/V [B*H, S, 64] bf16.
__global__ __launch_bounds__(128) void qkv_gemm_kernel(
    const __bf16* __restrict__ hb, const __bf16* __restrict__ wqb,
    __bf16* __restrict__ Qb, __bf16* __restrict__ Kb, __bf16* __restrict__ Vb) {
    const int lane  = threadIdx.x & 31;
    const int wave  = threadIdx.x >> 5;
    const int n0    = blockIdx.x * 64;                 // 48 groups of 4 N-tiles
    const int m0    = (blockIdx.y * 4 + wave) * 16;    // 0..511 tiles
    const int kbase = (lane >> 4) * 8;
    const int rowA  = m0 + (lane & 15);

    const __bf16* pa    = hb + (size_t)rowA * T5_DM;
    const __bf16* pbcol = wqb + (size_t)(n0 + (lane & 15)) * T5_DM + (lane >> 4) * 16;

    v8f c[4] = {};
    #pragma unroll 2
    for (int kk = 0; kk < T5_DM; kk += 32) {
        v8bf alo = *(const v8bf*)(pa + kk + (lane >> 4) * 16);
        v8bf ahi = *(const v8bf*)(pa + kk + (lane >> 4) * 16 + 8);
        v16bf a;
        #pragma unroll
        for (int i = 0; i < 8; ++i) { a[i] = alo[i]; a[8 + i] = ahi[i]; }
        #pragma unroll
        for (int j = 0; j < 4; ++j) {
            v16bf b = *(const v16bf*)(pbcol + (size_t)j * 16 * T5_DM + kk);
            c[j] = wmma_bf16(a, b, c[j]);
        }
    }

    #pragma unroll
    for (int j = 0; j < 4; ++j) {
        const int coln  = n0 + j * 16 + (lane & 15);
        const int which = coln >> 10;          // 0:q 1:k 2:v
        const int inner = coln & 1023;
        const int h  = inner >> 6;
        const int dd = inner & 63;
        __bf16* dst = (which == 0) ? Qb : ((which == 1) ? Kb : Vb);
        #pragma unroll
        for (int r = 0; r < 8; ++r) {
            int m  = m0 + r + kbase;           // kbase == (lane>>4)*8
            int bb = m >> 10;
            int ss = m & 1023;
            dst[(((size_t)(bb * T5_H + h) * T5_S + ss) << 6) + dd] = (__bf16)c[j][r];
        }
    }
}

// ------------------------- flash attention ----------------------------------
// One WG = one (b,h) x 64 q rows; 4 waves x 16 q rows; stream K/V in 32-row
// chunks via LDS; transposed-score trick keeps softmax in-lane (+1 shfl_xor16).
__global__ __launch_bounds__(128) void flash_attn_kernel(
    const __bf16* __restrict__ Qb, const __bf16* __restrict__ Kb,
    const __bf16* __restrict__ Vb, const float* __restrict__ biasTab,
    __bf16* __restrict__ attnb) {
    __shared__ float  sBias[2048];
    __shared__ __bf16 sK[32][64];     // [k][d] chunk
    __shared__ __bf16 sVt[64][32];    // [d][k] chunk (transposed)

    const int tid  = threadIdx.x;
    const int lane = tid & 31;
    const int wave = tid >> 5;
    const int bh   = blockIdx.x >> 4;            // b*16 + h
    const int qblk = blockIdx.x & 15;
    const int h    = bh & 15;
    const int b    = bh >> 4;
    const int q0    = qblk * 64 + wave * 16;
    const int kbase = (lane >> 4) * 8;
    const int qg    = q0 + (lane & 15);

    for (int i = tid; i < 2048; i += 128) sBias[i] = biasTab[h * 2048 + i];

    // Q tile as two B-operands (contiguous 32B per lane).
    const __bf16* pq = Qb + ((size_t)bh * T5_S + qg) * T5_DKV + (lane >> 4) * 16;
    const v16bf bq0 = *(const v16bf*)(pq);
    const v16bf bq1 = *(const v16bf*)(pq + 32);

    float m_run = -3.0e38f;
    float l_run = 0.0f;
    v8f acc[4] = {};

    const __bf16* kSrc = Kb + (size_t)bh * T5_S * T5_DKV;
    const __bf16* vSrc = Vb + (size_t)bh * T5_S * T5_DKV;

    for (int k0 = 0; k0 < T5_S; k0 += 32) {
        __syncthreads();
        // Stage K [32][64] (async global->LDS when available) and V transposed.
        for (int t = tid; t < 256; t += 128) {
            int kk = t >> 3, grp = t & 7;
            const __bf16* gk = kSrc + (size_t)(k0 + kk) * 64 + grp * 8;
#if defined(USE_ASYNC_LDS)
            __builtin_amdgcn_global_load_async_to_lds_b128(
                (__attribute__((address_space(1))) v4i*)gk,
                (__attribute__((address_space(3))) v4i*)&sK[kk][grp * 8],
                0, 0);
#else
            *(v8bf*)(&sK[kk][grp * 8]) = *(const v8bf*)gk;
#endif
            v8bf vv = *(const v8bf*)(vSrc + (size_t)(k0 + kk) * 64 + grp * 8);
            #pragma unroll
            for (int e = 0; e < 8; ++e) sVt[grp * 8 + e][kk] = vv[e];
        }
        wait_async_zero();
        __syncthreads();

        // Transposed scores: S_T[k, q] = sum_d K[k,d] * Q[q,d], two 16-k tiles.
        float p[16];
        #pragma unroll
        for (int st = 0; st < 2; ++st) {
            int krow = st * 16 + (lane & 15);
            v8bf t0 = *(const v8bf*)(&sK[krow][(lane >> 4) * 16]);
            v8bf t1 = *(const v8bf*)(&sK[krow][(lane >> 4) * 16 + 8]);
            v8bf t2 = *(const v8bf*)(&sK[krow][(lane >> 4) * 16 + 32]);
            v8bf t3 = *(const v8bf*)(&sK[krow][(lane >> 4) * 16 + 40]);
            v16bf a0, a1;
            #pragma unroll
            for (int i = 0; i < 8; ++i) {
                a0[i] = t0[i]; a0[8 + i] = t1[i];
                a1[i] = t2[i]; a1[8 + i] = t3[i];
            }
            v8f c = {};
            c = wmma_bf16(a0, bq0, c);
            c = wmma_bf16(a1, bq1, c);
            #pragma unroll
            for (int r = 0; r < 8; ++r) {
                int kg = k0 + st * 16 + r + kbase;
                p[st * 8 + r] = c[r] + sBias[kg - qg + 1023];
            }
        }

        // Online softmax (per q column; lanes l and l+16 share one q).
        float cmax = p[0];
        #pragma unroll
        for (int i = 1; i < 16; ++i) cmax = fmaxf(cmax, p[i]);
        cmax = fmaxf(cmax, __shfl_xor(cmax, 16, 32));
        float newm = fmaxf(m_run, cmax);
        float corr = __expf(m_run - newm);
        float psum = 0.0f;
        #pragma unroll
        for (int i = 0; i < 16; ++i) { p[i] = __expf(p[i] - newm); psum += p[i]; }
        psum += __shfl_xor(psum, 16, 32);
        l_run = l_run * corr + psum;
        m_run = newm;
        #pragma unroll
        for (int dt = 0; dt < 4; ++dt)
            #pragma unroll
            for (int r = 0; r < 8; ++r) acc[dt][r] *= corr;

        // Repack P_T into a WMMA B-operand (needs the partner half's k rows).
        float px0[8], px1[8];
        #pragma unroll
        for (int r = 0; r < 8; ++r) {
            px0[r] = __shfl_xor(p[r], 16, 32);
            px1[r] = __shfl_xor(p[8 + r], 16, 32);
        }
        const bool lo = (lane < 16);
        v16bf pbop;
        #pragma unroll
        for (int i = 0; i < 8; ++i) {
            pbop[i]     = (__bf16)(lo ? p[i]   : px1[i]);
            pbop[8 + i] = (__bf16)(lo ? px0[i] : p[8 + i]);
        }

        // out_T[d, q] += V_T[d, k] * P_T[k, q]  (4 d-tiles of 16)
        #pragma unroll
        for (int dt = 0; dt < 4; ++dt) {
            int drow = dt * 16 + (lane & 15);
            v8bf v0 = *(const v8bf*)(&sVt[drow][(lane >> 4) * 16]);
            v8bf v1 = *(const v8bf*)(&sVt[drow][(lane >> 4) * 16 + 8]);
            v16bf av;
            #pragma unroll
            for (int i = 0; i < 8; ++i) { av[i] = v0[i]; av[8 + i] = v1[i]; }
            acc[dt] = wmma_bf16(av, pbop, acc[dt]);
        }
    }

    const float inv = 1.0f / l_run;
    #pragma unroll
    for (int dt = 0; dt < 4; ++dt) {
        #pragma unroll
        for (int r = 0; r < 8; ++r) {
            int d = dt * 16 + r + kbase;
            attnb[((size_t)b * T5_S + qg) * T5_INNER + h * 64 + d] =
                (__bf16)(acc[dt][r] * inv);
        }
    }
}

// ------------------------- output projection GEMM ---------------------------
// out[m, j] = sum_i attn[m, i] * w_out[j, i]; M=8192, N=1024, K=1024. fp32 out.
// One wave -> 16x64 tile (4 accumulators, shared A operand).
__global__ __launch_bounds__(128) void out_gemm_kernel(
    const __bf16* __restrict__ ab, const __bf16* __restrict__ wob,
    float* __restrict__ out) {
    const int lane  = threadIdx.x & 31;
    const int wave  = threadIdx.x >> 5;
    const int n0    = blockIdx.x * 64;                 // 16 groups of 4 N-tiles
    const int m0    = (blockIdx.y * 4 + wave) * 16;
    const int kbase = (lane >> 4) * 8;
    const int rowA  = m0 + (lane & 15);

    const __bf16* pa    = ab + (size_t)rowA * T5_INNER;
    const __bf16* pbcol = wob + (size_t)(n0 + (lane & 15)) * T5_INNER + (lane >> 4) * 16;

    v8f c[4] = {};
    #pragma unroll 2
    for (int kk = 0; kk < T5_INNER; kk += 32) {
        v8bf alo = *(const v8bf*)(pa + kk + (lane >> 4) * 16);
        v8bf ahi = *(const v8bf*)(pa + kk + (lane >> 4) * 16 + 8);
        v16bf a;
        #pragma unroll
        for (int i = 0; i < 8; ++i) { a[i] = alo[i]; a[8 + i] = ahi[i]; }
        #pragma unroll
        for (int j = 0; j < 4; ++j) {
            v16bf b = *(const v16bf*)(pbcol + (size_t)j * 16 * T5_INNER + kk);
            c[j] = wmma_bf16(a, b, c[j]);
        }
    }
    #pragma unroll
    for (int j = 0; j < 4; ++j) {
        const int coln = n0 + j * 16 + (lane & 15);
        #pragma unroll
        for (int r = 0; r < 8; ++r) {
            int m = m0 + r + kbase;
            out[(size_t)m * T5_DM + coln] = c[j][r];
        }
    }
}

// ---------------------------------------------------------------------------
extern "C" void kernel_launch(void* const* d_in, const int* in_sizes, int n_in,
                              void* d_out, int out_size, void* d_ws, size_t ws_size,
                              hipStream_t stream) {
    const float* hidden  = (const float*)d_in[0];  // [8,1024,1024]
    const float* w_qkv   = (const float*)d_in[1];  // [3072,1024]
    const float* rel_emb = (const float*)d_in[2];  // [32,16]
    const float* w_out   = (const float*)d_in[3];  // [1024,1024]
    float* out = (float*)d_out;                    // [8,1024,1024]

    // Workspace carving (all sizes 256B-aligned multiples).
    char* w = (char*)d_ws;
    __bf16* hb  = (__bf16*)w; w += (size_t)T5_B * T5_S * T5_DM * 2;        // 16 MB
    __bf16* wqb = (__bf16*)w; w += (size_t)3 * T5_INNER * T5_DM * 2;       //  6 MB
    __bf16* wob = (__bf16*)w; w += (size_t)T5_DM * T5_INNER * 2;           //  2 MB
    __bf16* Qb  = (__bf16*)w; w += (size_t)T5_B * T5_H * T5_S * T5_DKV * 2;// 16 MB
    __bf16* Kb  = (__bf16*)w; w += (size_t)T5_B * T5_H * T5_S * T5_DKV * 2;// 16 MB
    __bf16* Vb  = (__bf16*)w; w += (size_t)T5_B * T5_H * T5_S * T5_DKV * 2;// 16 MB
    float*  biasTab = (float*)w; w += (size_t)T5_H * 2048 * 4;             // 128 KB
    __bf16* attnb   = (__bf16*)w;                                          // 16 MB

    // 1) fp32 -> bf16 conversions
    {
        int n4 = (T5_B * T5_S * T5_DM) / 4;
        cvt_f32_bf16_kernel<<<(n4 + 255) / 256, 256, 0, stream>>>(hidden, hb, n4);
    }
    {
        int n4 = (3 * T5_INNER * T5_DM) / 4;
        cvt_f32_bf16_kernel<<<(n4 + 255) / 256, 256, 0, stream>>>(w_qkv, wqb, n4);
    }
    {
        int n4 = (T5_DM * T5_INNER) / 4;
        cvt_f32_bf16_kernel<<<(n4 + 255) / 256, 256, 0, stream>>>(w_out, wob, n4);
    }

    // 2) relative-position bias table [16][2048]
    bias_table_kernel<<<(T5_H * 2048 + 255) / 256, 256, 0, stream>>>(rel_emb, biasTab);

    // 3) QKV projection: grid = (48 N-groups, 128 M-groups), 4 waves/WG
    qkv_gemm_kernel<<<dim3(48, 128), 128, 0, stream>>>(hb, wqb, Qb, Kb, Vb);

    // 4) flash attention: one WG per (b,h,64-q block) -> 8*16*16 = 2048 WGs
    flash_attn_kernel<<<2048, 128, 0, stream>>>(Qb, Kb, Vb, biasTab, attnb);

    // 5) output projection: grid = (16, 128)
    out_gemm_kernel<<<dim3(16, 128), 128, 0, stream>>>(attnb, wob, out);
}
